// LinearQwen3NextSparseMoeBlock_7687991460319
// MI455X (gfx1250) — matrix operand order repaired and checked by reading, not compiled, not run on hardware
//
#include <hip/hip_runtime.h>
#include <hip/hip_bf16.h>

typedef __attribute__((ext_vector_type(16))) __bf16 v16bf;
typedef __attribute__((ext_vector_type(8)))  float  v8f;
typedef __attribute__((ext_vector_type(4)))  int    v4i;

static constexpr int TOK  = 8192;   // B*S
static constexpr int DIM  = 1024;   // D
static constexpr int NEXP = 16;     // experts
static constexpr int FI   = 512;    // expert intermediate
static constexpr int FSH  = 2048;   // shared intermediate

union Frag16 { v16bf v; float4 f[2]; };

// ---- gfx1250 async global->LDS copy (ASYNCcnt) with safe fallback ----------
#if defined(__has_builtin)
#  if __has_builtin(__builtin_amdgcn_global_load_async_to_lds_b128)
#    define HAS_ASYNC_COPY 1
#  endif
#endif
#ifndef HAS_ASYNC_COPY
#  define HAS_ASYNC_COPY 0
#endif

#if HAS_ASYNC_COPY
typedef __attribute__((address_space(1))) v4i gv4i;   // global 128-bit chunks
typedef __attribute__((address_space(3))) v4i lv4i;   // LDS 128-bit chunks
__device__ __forceinline__ void async_cp16(void* lds, const void* gmem) {
  __builtin_amdgcn_global_load_async_to_lds_b128((gv4i*)gmem, (lv4i*)lds, 0, 0);
}
#if __has_builtin(__builtin_amdgcn_s_wait_asynccnt)
__device__ __forceinline__ void wait_async() { __builtin_amdgcn_s_wait_asynccnt(0); }
#else
__device__ __forceinline__ void wait_async() { asm volatile("s_wait_asynccnt 0" ::: "memory"); }
#endif
#endif

__device__ __forceinline__ float fast_sigmoid(float v) {
  return __builtin_amdgcn_rcpf(1.f + __expf(-v));
}

// ---------------------------------------------------------------- utilities
__global__ void k_init(int* cnt) {
  if (threadIdx.x < NEXP) cnt[threadIdx.x] = 0;
}

__global__ void k_scan(const int* __restrict__ cnt, int* __restrict__ offs) {
  if (threadIdx.x == 0) {
    int s = 0;
    for (int e = 0; e < NEXP; ++e) { offs[e] = s; s += cnt[e]; }
  }
}

__global__ void __launch_bounds__(256) k_cvt_bf16(const float* __restrict__ src,
                                                  __bf16* __restrict__ dst, int n4) {
  int i = blockIdx.x * blockDim.x + threadIdx.x;
  if (i >= n4) return;
  float4 v = ((const float4*)src)[i];
  union { __bf16 h[4]; uint2 u; } o;
  o.h[0] = (__bf16)v.x; o.h[1] = (__bf16)v.y; o.h[2] = (__bf16)v.z; o.h[3] = (__bf16)v.w;
  ((uint2*)dst)[i] = o.u;
}

// src [batch][R][C] f32  ->  dst [batch][C][R] bf16   (R, C multiples of 32)
__global__ void __launch_bounds__(256) k_transpose(const float* __restrict__ src,
                                                   __bf16* __restrict__ dst,
                                                   int R, int C) {
  __shared__ float tile[32][33];
  size_t bofs = (size_t)blockIdx.z * R * C;
  src += bofs; dst += bofs;
  int c0 = blockIdx.x * 32, r0 = blockIdx.y * 32;
  #pragma unroll
  for (int i = 0; i < 4; ++i) {
    int r = r0 + threadIdx.y + i * 8;
    tile[threadIdx.y + i * 8][threadIdx.x] = src[(size_t)r * C + c0 + threadIdx.x];
  }
  __syncthreads();
  #pragma unroll
  for (int i = 0; i < 4; ++i) {
    int c = c0 + threadIdx.y + i * 8;
    dst[(size_t)c * R + r0 + threadIdx.x] = (__bf16)tile[threadIdx.x][threadIdx.y + i * 8];
  }
}

// ---------------------------------------------------------------- router
__global__ void __launch_bounds__(256) k_router(const float* __restrict__ x,
                                                const float* __restrict__ rw,
                                                const float* __restrict__ sgw,
                                                int* __restrict__ cnt,
                                                int* __restrict__ slot_tok,
                                                float* __restrict__ slot_w,
                                                float* __restrict__ gate) {
  int t = blockIdx.x * blockDim.x + threadIdx.x;
  if (t >= TOK) return;
  const float* xr = x + (size_t)t * DIM;

  float logits[NEXP];
  for (int e = 0; e < NEXP; ++e) {
    const float* w = rw + (size_t)e * DIM;
    float acc = 0.f;
    for (int d = 0; d < DIM; d += 4) {
      float4 a = *(const float4*)(xr + d);
      float4 b = *(const float4*)(w + d);
      acc = fmaf(a.x, b.x, acc); acc = fmaf(a.y, b.y, acc);
      acc = fmaf(a.z, b.z, acc); acc = fmaf(a.w, b.w, acc);
    }
    logits[e] = acc;
  }
  float m = logits[0];
  #pragma unroll
  for (int e = 1; e < NEXP; ++e) m = fmaxf(m, logits[e]);
  float p[NEXP];
  #pragma unroll
  for (int e = 0; e < NEXP; ++e) p[e] = __expf(logits[e] - m);
  // top-2 (softmax denominator cancels under renorm)
  int i0 = 0;
  #pragma unroll
  for (int e = 1; e < NEXP; ++e) if (p[e] > p[i0]) i0 = e;
  int i1 = (i0 == 0) ? 1 : 0;
  #pragma unroll
  for (int e = 0; e < NEXP; ++e) if (e != i0 && p[e] > p[i1]) i1 = e;
  float w0 = p[i0], w1 = p[i1];
  float inv = __builtin_amdgcn_rcpf(w0 + w1);
  w0 *= inv; w1 *= inv;

  int p0 = atomicAdd(cnt + i0, 1);
  slot_tok[(size_t)i0 * TOK + p0] = t;
  slot_w  [(size_t)i0 * TOK + p0] = w0;
  int p1 = atomicAdd(cnt + i1, 1);
  slot_tok[(size_t)i1 * TOK + p1] = t;
  slot_w  [(size_t)i1 * TOK + p1] = w1;

  float ga = 0.f;
  for (int d = 0; d < DIM; d += 4) {
    float4 a = *(const float4*)(xr + d);
    float4 b = *(const float4*)(sgw + d);
    ga = fmaf(a.x, b.x, ga); ga = fmaf(a.y, b.y, ga);
    ga = fmaf(a.z, b.z, ga); ga = fmaf(a.w, b.w, ga);
  }
  gate[t] = fast_sigmoid(ga);
}

// ---------------------------------------------------------------- gate+up (SwiGLU) GEMM
// Block tile 64(M) x 128(N), 8 waves as 2x4, each wave 32x32 (2x2 WMMA subtiles),
// double-buffered LDS with async global->LDS producers, one barrier per K-step.
template <bool GATHER>
__global__ void __launch_bounds__(256) k_gateup(const __bf16* __restrict__ X,    // [TOK, Kd]
                                                const __bf16* __restrict__ BgT,  // [(E)][Nfull, Kd]
                                                const __bf16* __restrict__ BuT,
                                                __bf16* __restrict__ H,          // [*, Nfull]
                                                const int* __restrict__ cnt,
                                                const int* __restrict__ offs,
                                                const int* __restrict__ slot_tok,
                                                int Kd, int Nfull) {
  int e      = GATHER ? blockIdx.z : 0;
  int rows   = GATHER ? cnt[e] : TOK;
  int m_base = blockIdx.x * 64;
  if (m_base >= rows) return;
  int n_base = blockIdx.y * 128;
  size_t wstride = (size_t)Nfull * Kd;
  const __bf16* Bg = BgT + (size_t)e * wstride;
  const __bf16* Bu = BuT + (size_t)e * wstride;
  int hbase = (GATHER ? offs[e] : 0) + m_base;

  __shared__ __align__(16) __bf16 lA [2][64 * 32];
  __shared__ __align__(16) __bf16 lBg[2][128 * 32];
  __shared__ __align__(16) __bf16 lBu[2][128 * 32];
  __shared__ int ltok[64];

  int tid = threadIdx.x;
  if (tid < 64) {
    int rr = m_base + tid;
    int rc = rr < rows ? rr : rows - 1;
    ltok[tid] = GATHER ? slot_tok[(size_t)e * TOK + rc] : rc;
  }
  __syncthreads();

  int lane = tid & 31, wv = tid >> 5;
  int wm = wv >> 2, wn = wv & 3;
  int hsel = lane >> 4, l15 = lane & 15;

  // loader assignment: A 1 x b128 / thread / step; Bg,Bu 2 x b128 each
  int arow  = ltok[tid >> 2];
  int aOff  = (tid >> 2) * 32 + (tid & 3) * 8;
  int bOff  = (tid >> 1) * 32 + (tid & 1) * 16;
  const __bf16* aSrc = X  + (size_t)arow * Kd + (tid & 3) * 8;
  const __bf16* gSrc = Bg + (size_t)(n_base + (tid >> 1)) * Kd + (tid & 1) * 16;
  const __bf16* uSrc = Bu + (size_t)(n_base + (tid >> 1)) * Kd + (tid & 1) * 16;

  // stage K-tile 0 into buffer 0
#if HAS_ASYNC_COPY
  async_cp16(lA [0] + aOff,     aSrc);
  async_cp16(lBg[0] + bOff,     gSrc);
  async_cp16(lBg[0] + bOff + 8, gSrc + 8);
  async_cp16(lBu[0] + bOff,     uSrc);
  async_cp16(lBu[0] + bOff + 8, uSrc + 8);
  wait_async();
#else
  {
    float4 ra  = *(const float4*)aSrc;
    float4 rg0 = *(const float4*)gSrc;
    float4 rg1 = *(const float4*)(gSrc + 8);
    float4 ru0 = *(const float4*)uSrc;
    float4 ru1 = *(const float4*)(uSrc + 8);
    *(float4*)(lA [0] + aOff)     = ra;
    *(float4*)(lBg[0] + bOff)     = rg0;
    *(float4*)(lBg[0] + bOff + 8) = rg1;
    *(float4*)(lBu[0] + bOff)     = ru0;
    *(float4*)(lBu[0] + bOff + 8) = ru1;
  }
#endif
  __syncthreads();

  v8f accG[2][2] = {}, accU[2][2] = {};
  int cur = 0;

  for (int k0 = 0; k0 < Kd; k0 += 32) {
    bool has_next = (k0 + 32) < Kd;
#if HAS_ASYNC_COPY
    if (has_next) {
      int nb = cur ^ 1;
      async_cp16(lA [nb] + aOff,     aSrc + k0 + 32);
      async_cp16(lBg[nb] + bOff,     gSrc + k0 + 32);
      async_cp16(lBg[nb] + bOff + 8, gSrc + k0 + 40);
      async_cp16(lBu[nb] + bOff,     uSrc + k0 + 32);
      async_cp16(lBu[nb] + bOff + 8, uSrc + k0 + 40);
    }
#else
    float4 ra, rg0, rg1, ru0, ru1;
    if (has_next) {
      ra  = *(const float4*)(aSrc + k0 + 32);
      rg0 = *(const float4*)(gSrc + k0 + 32);
      rg1 = *(const float4*)(gSrc + k0 + 40);
      ru0 = *(const float4*)(uSrc + k0 + 32);
      ru1 = *(const float4*)(uSrc + k0 + 40);
    }
#endif

    const __bf16* lAc = lA[cur];
    const __bf16* lGc = lBg[cur];
    const __bf16* lUc = lBu[cur];
    Frag16 a[2], bg[2], bu[2];
    #pragma unroll
    for (int mi = 0; mi < 2; ++mi) {
      const __bf16* pa = lAc + (wm * 32 + mi * 16 + l15) * 32 + hsel * 8;
      a[mi].f[0] = *(const float4*)pa;
      a[mi].f[1] = *(const float4*)(pa + 16);
    }
    #pragma unroll
    for (int ni = 0; ni < 2; ++ni) {
      const __bf16* pg = lGc + (wn * 32 + ni * 16 + l15) * 32 + hsel * 8;
      bg[ni].f[0] = *(const float4*)pg;
      bg[ni].f[1] = *(const float4*)(pg + 16);
      const __bf16* pu = lUc + (wn * 32 + ni * 16 + l15) * 32 + hsel * 8;
      bu[ni].f[0] = *(const float4*)pu;
      bu[ni].f[1] = *(const float4*)(pu + 16);
    }
    #pragma unroll
    for (int mi = 0; mi < 2; ++mi)
      #pragma unroll
      for (int ni = 0; ni < 2; ++ni) {
        accG[mi][ni] = __builtin_amdgcn_wmma_f32_16x16x32_bf16(
            false, a[mi].v, false, bg[ni].v, (short)0, accG[mi][ni], false, false);
        accU[mi][ni] = __builtin_amdgcn_wmma_f32_16x16x32_bf16(
            false, a[mi].v, false, bu[ni].v, (short)0, accU[mi][ni], false, false);
      }

#if HAS_ASYNC_COPY
    wait_async();
#else
    if (has_next) {
      int nb = cur ^ 1;
      *(float4*)(lA [nb] + aOff)     = ra;
      *(float4*)(lBg[nb] + bOff)     = rg0;
      *(float4*)(lBg[nb] + bOff + 8) = rg1;
      *(float4*)(lBu[nb] + bOff)     = ru0;
      *(float4*)(lBu[nb] + bOff + 8) = ru1;
    }
#endif
    __syncthreads();
    cur ^= 1;
  }

  #pragma unroll
  for (int mi = 0; mi < 2; ++mi)
    #pragma unroll
    for (int ni = 0; ni < 2; ++ni) {
      int col = n_base + wn * 32 + ni * 16 + l15;
      #pragma unroll
      for (int v = 0; v < 8; ++v) {
        int rl = wm * 32 + mi * 16 + hsel * 8 + v;
        if (m_base + rl < rows) {
          float g = accG[mi][ni][v], u = accU[mi][ni][v];
          float h = g * u * fast_sigmoid(g);   // silu(g)*u
          H[(size_t)(hbase + rl) * Nfull + col] = (__bf16)h;
        }
      }
    }
}

// ---------------------------------------------------------------- down-proj GEMM
// Same 64x128 tiling. GATHER: out[tok,n] += w_slot * (H@Bw^T) (atomic, runs last);
// !GATHER: out[t,n] = gate[t] * (H@Bw^T) (plain store, runs first).
template <bool GATHER>
__global__ void __launch_bounds__(256) k_down(const __bf16* __restrict__ Hin,  // [*, Kd]
                                              const __bf16* __restrict__ BwT,  // [(E)][DIM, Kd]
                                              float* __restrict__ out,         // [TOK, DIM]
                                              const int* __restrict__ cnt,
                                              const int* __restrict__ offs,
                                              const int* __restrict__ slot_tok,
                                              const float* __restrict__ slot_w,
                                              const float* __restrict__ gate,
                                              int Kd) {
  int e      = GATHER ? blockIdx.z : 0;
  int rows   = GATHER ? cnt[e] : TOK;
  int m_base = blockIdx.x * 64;
  if (m_base >= rows) return;
  int n_base = blockIdx.y * 128;
  const __bf16* Bw = BwT + (size_t)e * DIM * Kd;
  int abase = (GATHER ? offs[e] : 0) + m_base;

  __shared__ __align__(16) __bf16 lA[2][64 * 32];
  __shared__ __align__(16) __bf16 lB[2][128 * 32];
  __shared__ int   ltok[64];
  __shared__ float lw[64];

  int tid = threadIdx.x;
  if (tid < 64) {
    int rr = m_base + tid;
    int rc = rr < rows ? rr : rows - 1;
    if (GATHER) {
      ltok[tid] = slot_tok[(size_t)e * TOK + rc];
      lw[tid]   = slot_w  [(size_t)e * TOK + rc];
    } else {
      ltok[tid] = rc;
      lw[tid]   = gate[rc];
    }
  }
  __syncthreads();

  int lane = tid & 31, wv = tid >> 5;
  int wm = wv >> 2, wn = wv & 3;
  int hsel = lane >> 4, l15 = lane & 15;

  int aOff = (tid >> 2) * 32 + (tid & 3) * 8;
  int bOff = (tid >> 1) * 32 + (tid & 1) * 16;
  const __bf16* aSrc = Hin + (size_t)(abase + (tid >> 2)) * Kd + (tid & 3) * 8;
  const __bf16* bSrc = Bw  + (size_t)(n_base + (tid >> 1)) * Kd + (tid & 1) * 16;

#if HAS_ASYNC_COPY
  async_cp16(lA[0] + aOff,     aSrc);
  async_cp16(lB[0] + bOff,     bSrc);
  async_cp16(lB[0] + bOff + 8, bSrc + 8);
  wait_async();
#else
  {
    float4 ra  = *(const float4*)aSrc;
    float4 rb0 = *(const float4*)bSrc;
    float4 rb1 = *(const float4*)(bSrc + 8);
    *(float4*)(lA[0] + aOff)     = ra;
    *(float4*)(lB[0] + bOff)     = rb0;
    *(float4*)(lB[0] + bOff + 8) = rb1;
  }
#endif
  __syncthreads();

  v8f acc[2][2] = {};
  int cur = 0;

  for (int k0 = 0; k0 < Kd; k0 += 32) {
    bool has_next = (k0 + 32) < Kd;
#if HAS_ASYNC_COPY
    if (has_next) {
      int nb = cur ^ 1;
      async_cp16(lA[nb] + aOff,     aSrc + k0 + 32);
      async_cp16(lB[nb] + bOff,     bSrc + k0 + 32);
      async_cp16(lB[nb] + bOff + 8, bSrc + k0 + 40);
    }
#else
    float4 ra, rb0, rb1;
    if (has_next) {
      ra  = *(const float4*)(aSrc + k0 + 32);
      rb0 = *(const float4*)(bSrc + k0 + 32);
      rb1 = *(const float4*)(bSrc + k0 + 40);
    }
#endif

    const __bf16* lAc = lA[cur];
    const __bf16* lBc = lB[cur];
    Frag16 a[2], b[2];
    #pragma unroll
    for (int mi = 0; mi < 2; ++mi) {
      const __bf16* pa = lAc + (wm * 32 + mi * 16 + l15) * 32 + hsel * 8;
      a[mi].f[0] = *(const float4*)pa;
      a[mi].f[1] = *(const float4*)(pa + 16);
    }
    #pragma unroll
    for (int ni = 0; ni < 2; ++ni) {
      const __bf16* pb = lBc + (wn * 32 + ni * 16 + l15) * 32 + hsel * 8;
      b[ni].f[0] = *(const float4*)pb;
      b[ni].f[1] = *(const float4*)(pb + 16);
    }
    #pragma unroll
    for (int mi = 0; mi < 2; ++mi)
      #pragma unroll
      for (int ni = 0; ni < 2; ++ni)
        acc[mi][ni] = __builtin_amdgcn_wmma_f32_16x16x32_bf16(
            false, a[mi].v, false, b[ni].v, (short)0, acc[mi][ni], false, false);

#if HAS_ASYNC_COPY
    wait_async();
#else
    if (has_next) {
      int nb = cur ^ 1;
      *(float4*)(lA[nb] + aOff)     = ra;
      *(float4*)(lB[nb] + bOff)     = rb0;
      *(float4*)(lB[nb] + bOff + 8) = rb1;
    }
#endif
    __syncthreads();
    cur ^= 1;
  }

  #pragma unroll
  for (int mi = 0; mi < 2; ++mi)
    #pragma unroll
    for (int ni = 0; ni < 2; ++ni) {
      int col = n_base + wn * 32 + ni * 16 + l15;
      #pragma unroll
      for (int v = 0; v < 8; ++v) {
        int rl = wm * 32 + mi * 16 + hsel * 8 + v;
        if (m_base + rl < rows) {
          float val = acc[mi][ni][v] * lw[rl];
          if (GATHER) atomicAdd(out + (size_t)ltok[rl] * DIM + col, val);
          else        out[(size_t)ltok[rl] * DIM + col] = val;
        }
      }
    }
}

// ---------------------------------------------------------------- host launch
extern "C" void kernel_launch(void* const* d_in, const int* in_sizes, int n_in,
                              void* d_out, int out_size, void* d_ws, size_t ws_size,
                              hipStream_t stream) {
  const float* x        = (const float*)d_in[0];  // [T, D]
  const float* router_w = (const float*)d_in[1];  // [E, D]
  const float* Wg       = (const float*)d_in[2];  // [E, D, F]
  const float* Wu       = (const float*)d_in[3];  // [E, D, F]
  const float* Wd       = (const float*)d_in[4];  // [E, F, D]
  const float* Sg       = (const float*)d_in[5];  // [D, FS]
  const float* Su       = (const float*)d_in[6];  // [D, FS]
  const float* Sd       = (const float*)d_in[7];  // [FS, D]
  const float* sgw      = (const float*)d_in[8];  // [D, 1]
  float* out            = (float*)d_out;          // [T, D]

  char* ws = (char*)d_ws;
  size_t off = 0;
  auto alloc = [&](size_t bytes) -> void* {
    void* p = ws + off;
    off += (bytes + 255) & ~(size_t)255;
    return p;
  };

  __bf16* Xb   = (__bf16*)alloc((size_t)TOK * DIM * 2);
  __bf16* WgT  = (__bf16*)alloc((size_t)NEXP * FI * DIM * 2);   // [E][F][D]
  __bf16* WuT  = (__bf16*)alloc((size_t)NEXP * FI * DIM * 2);
  __bf16* WdT  = (__bf16*)alloc((size_t)NEXP * DIM * FI * 2);   // [E][D][F]
  __bf16* SgT  = (__bf16*)alloc((size_t)FSH * DIM * 2);         // [FS][D]
  __bf16* SuT  = (__bf16*)alloc((size_t)FSH * DIM * 2);
  __bf16* SdT  = (__bf16*)alloc((size_t)DIM * FSH * 2);         // [D][FS]
  __bf16* Hbuf = (__bf16*)alloc((size_t)TOK * FSH * 2);         // shared H, reused as expert H (2T x F)
  int*   cnt      = (int*)alloc(NEXP * 4);
  int*   offs     = (int*)alloc(NEXP * 4);
  int*   slot_tok = (int*)alloc((size_t)NEXP * TOK * 4);
  float* slot_w   = (float*)alloc((size_t)NEXP * TOK * 4);
  float* gatev    = (float*)alloc((size_t)TOK * 4);
  (void)ws_size; (void)in_sizes; (void)n_in; (void)out_size;

  // ---- prep: counters, bf16 conversion, weight transposes to [out, in]
  k_init<<<1, 32, 0, stream>>>(cnt);

  int n4 = TOK * DIM / 4;
  k_cvt_bf16<<<n4 / 256, 256, 0, stream>>>(x, Xb, n4);

  k_transpose<<<dim3(FI / 32, DIM / 32, NEXP), dim3(32, 8), 0, stream>>>(Wg, WgT, DIM, FI);
  k_transpose<<<dim3(FI / 32, DIM / 32, NEXP), dim3(32, 8), 0, stream>>>(Wu, WuT, DIM, FI);
  k_transpose<<<dim3(DIM / 32, FI / 32, NEXP), dim3(32, 8), 0, stream>>>(Wd, WdT, FI, DIM);
  k_transpose<<<dim3(FSH / 32, DIM / 32, 1),   dim3(32, 8), 0, stream>>>(Sg, SgT, DIM, FSH);
  k_transpose<<<dim3(FSH / 32, DIM / 32, 1),   dim3(32, 8), 0, stream>>>(Su, SuT, DIM, FSH);
  k_transpose<<<dim3(DIM / 32, FSH / 32, 1),   dim3(32, 8), 0, stream>>>(Sd, SdT, FSH, DIM);

  // ---- router: top-2 buckets + sigmoid gate, then exclusive scan of counts
  k_router<<<TOK / 256, 256, 0, stream>>>(x, router_w, sgw, cnt, slot_tok, slot_w, gatev);
  k_scan<<<1, 1, 0, stream>>>(cnt, offs);

  // ---- shared expert first (plain stores fully initialize out)
  k_gateup<false><<<dim3(TOK / 64, FSH / 128, 1), 256, 0, stream>>>(
      Xb, SgT, SuT, Hbuf, cnt, offs, slot_tok, DIM, FSH);
  k_down<false><<<dim3(TOK / 64, DIM / 128, 1), 256, 0, stream>>>(
      Hbuf, SdT, out, cnt, offs, slot_tok, slot_w, gatev, FSH);

  // ---- routed experts (gathered rows, atomic f32 combine into out)
  k_gateup<true><<<dim3(TOK / 64, FI / 128, NEXP), 256, 0, stream>>>(
      Xb, WgT, WuT, Hbuf, cnt, offs, slot_tok, DIM, FI);
  k_down<true><<<dim3(TOK / 64, DIM / 128, NEXP), 256, 0, stream>>>(
      Hbuf, WdT, out, cnt, offs, slot_tok, slot_w, gatev, FI);
}
